// WindowAttention_72438918414529
// MI455X (gfx1250) — compile-verified, matrix operand order checked by reading, not run on hardware
//
#include <hip/hip_runtime.h>
#include <hip/hip_bf16.h>

// ---------------------------------------------------------------------------
// Window attention (seq = B = 4) for MI455X / gfx1250.
// bf16 WMMA GEMMs + GLOBAL_LOAD_ASYNC_TO_LDS double-buffered staging.
// Block tile 256x128, 8 waves, 64x64 wave tile (4x4 WMMA 16x16x32 subtiles).
// ---------------------------------------------------------------------------

typedef __attribute__((ext_vector_type(16))) __bf16 v16bf;
typedef __attribute__((ext_vector_type(8)))  float  v8f;

#define PSZ   4
#define NHEAD 16
#define NCH   32
#define EDIM  2048
#define HDIM  128           // EDIM / NHEAD
#define BSEQ  4             // batch == attention sequence length
#define NWIN  4096          // 16^3 windows
#define MROWS (BSEQ * NWIN) // 16384 token rows
#define K3E   (3 * EDIM)    // 6144

// GEMM tiling
#define BM    256
#define BN    128
#define BK    32
#define LDST  40            // padded LDS row stride in bf16 elements (80B = 20 banks)

// ---------------------------------------------------------------------------
// f32 -> bf16 elementwise convert
// ---------------------------------------------------------------------------
__global__ __launch_bounds__(256)
void cvt_bf16_kernel(const float* __restrict__ src, __bf16* __restrict__ dst, int n) {
  int i = blockIdx.x * blockDim.x + threadIdx.x;
  if (i < n) dst[i] = (__bf16)src[i];
}

// ---------------------------------------------------------------------------
// patchify: x (B,C,64,64,64) f32 -> tok (B*N, E) bf16
// e = c*64 + px*16 + py*4 + pz ; n = ((nx*16)+ny)*16 + nz
// ---------------------------------------------------------------------------
__global__ __launch_bounds__(256)
void patchify_kernel(const float* __restrict__ x, __bf16* __restrict__ tok) {
  int idx = blockIdx.x * blockDim.x + threadIdx.x;   // < 33,554,432
  int e = idx & (EDIM - 1);
  int n = (idx >> 11) & (NWIN - 1);
  int b = idx >> 23;
  int c  = e >> 6;
  int px = (e >> 4) & 3;
  int py = (e >> 2) & 3;
  int pz = e & 3;
  int nx = n >> 8;
  int ny = (n >> 4) & 15;
  int nz = n & 15;
  size_t xi = ((((size_t)(b * NCH + c) * 64 + (nx * 4 + px)) * 64 + (ny * 4 + py)) * 64)
              + (nz * 4 + pz);
  tok[idx] = (__bf16)x[xi];
}

// ---------------------------------------------------------------------------
// window reverse scatter: y2 (B*N, E) f32 -> out (B,C,64,64,64) f32
// out[b,c, px*16+nx, py*16+ny, pz*16+nz]
// ---------------------------------------------------------------------------
__global__ __launch_bounds__(256)
void scatter_kernel(const float* __restrict__ y2, float* __restrict__ out) {
  int idx = blockIdx.x * blockDim.x + threadIdx.x;
  int e = idx & (EDIM - 1);
  int n = (idx >> 11) & (NWIN - 1);
  int b = idx >> 23;
  int c  = e >> 6;
  int px = (e >> 4) & 3;
  int py = (e >> 2) & 3;
  int pz = e & 3;
  int nx = n >> 8;
  int ny = (n >> 4) & 15;
  int nz = n & 15;
  size_t oi = ((((size_t)(b * NCH + c) * 64 + (px * 16 + nx)) * 64 + (py * 16 + ny)) * 64)
              + (pz * 16 + nz);
  out[oi] = y2[idx];
}

// ---------------------------------------------------------------------------
// async 16B global -> LDS copy (CDNA5 GLOBAL_LOAD_ASYNC_TO_LDS_B128, ASYNCcnt)
// ldsAddr = LDS byte offset (low 32 bits of the flat shared pointer, ISA §10.2)
// ---------------------------------------------------------------------------
__device__ __forceinline__ void async_cp16(unsigned ldsAddr, const void* gptr) {
  asm volatile("global_load_async_to_lds_b128 %0, %1, off"
               :
               : "v"(ldsAddr), "v"((unsigned long long)gptr)
               : "memory");
}
__device__ __forceinline__ void wait_async0() {
  asm volatile("s_wait_asynccnt 0x0" ::: "memory");
}

// ---------------------------------------------------------------------------
// bf16 WMMA GEMM:  out[m,n] = sum_k A[m,k] * W[n,k] + bias[n]
// A: (M,K) bf16 row-major, W: (N,K) bf16 row-major (i.e. A @ W^T)
// ---------------------------------------------------------------------------
__global__ __launch_bounds__(256)
void gemm_bf16_wmma_kernel(const __bf16* __restrict__ A,
                           const __bf16* __restrict__ W,
                           const float*  __restrict__ bias,
                           void* __restrict__ out,
                           int M, int N, int K, int outF32)
{
  __shared__ __align__(16) __bf16 sA[2][BM * LDST];   // 2 x 20480 B
  __shared__ __align__(16) __bf16 sB[2][BN * LDST];   // 2 x 10240 B

  const int t      = threadIdx.x;
  const int blockM = blockIdx.y * BM;
  const int blockN = blockIdx.x * BN;

  // cooperative tile load mapping: 16B vectors, 64 rows apart per thread
  const int r0 = t >> 2;           // 0..63
  const int c0 = (t & 3) * 8;      // element column: 0,8,16,24

  // global source pointers (A: rows r0+{0,64,128,192}; B: rows r0+{0,64})
  const __bf16* aG[4];
  const __bf16* bG[2];
#pragma unroll
  for (int i = 0; i < 4; ++i) aG[i] = A + (size_t)(blockM + r0 + i * 64) * K + c0;
#pragma unroll
  for (int i = 0; i < 2; ++i) bG[i] = W + (size_t)(blockN + r0 + i * 64) * K + c0;

  // LDS byte addresses of this thread's destination slots, per buffer
  unsigned dA[2][4], dB[2][2];
#pragma unroll
  for (int bf = 0; bf < 2; ++bf) {
#pragma unroll
    for (int i = 0; i < 4; ++i)
      dA[bf][i] = (unsigned)(unsigned long long)&sA[bf][(r0 + i * 64) * LDST + c0];
#pragma unroll
    for (int i = 0; i < 2; ++i)
      dB[bf][i] = (unsigned)(unsigned long long)&sB[bf][(r0 + i * 64) * LDST + c0];
  }

  // wave / lane geometry: 8 waves = 4 (M) x 2 (N), each owns a 64x64 tile
  const int w       = t >> 5;
  const int lane    = t & 31;
  const int half    = lane >> 4;       // 0: lanes 0-15, 1: lanes 16-31
  const int lr      = lane & 15;
  const int kb      = half * 8;        // K base element for this half (ISA A/B layout)
  const int rowBase = (w >> 1) * 64;   // 4 waves along M
  const int colBase = (w & 1) * 64;    // 2 waves along N

  v8f acc[4][4];
  {
    v8f z = {};
#pragma unroll
    for (int mi = 0; mi < 4; ++mi)
#pragma unroll
      for (int ni = 0; ni < 4; ++ni) acc[mi][ni] = z;
  }

  union Frag { uint4 u[2]; v16bf v; };

  // stream one K-slab into LDS buffer `nb`
  auto issue = [&](int koElems, int nb) {
#pragma unroll
    for (int i = 0; i < 4; ++i) async_cp16(dA[nb][i], aG[i] + koElems);
#pragma unroll
    for (int i = 0; i < 2; ++i) async_cp16(dB[nb][i], bG[i] + koElems);
  };

  // 16 WMMAs on LDS buffer `cb` (call with literal cb -> immediate LDS offsets)
  auto compute = [&](int cb) {
    const __bf16* la = sA[cb];
    const __bf16* lb = sB[cb];
    Frag af[4], bf[4];
#pragma unroll
    for (int mi = 0; mi < 4; ++mi) {
      const int row = rowBase + mi * 16 + lr;
      af[mi].u[0] = *(const uint4*)&la[row * LDST + kb];        // K kb..kb+7
      af[mi].u[1] = *(const uint4*)&la[row * LDST + kb + 16];   // K kb+16..kb+23
    }
#pragma unroll
    for (int ni = 0; ni < 4; ++ni) {
      const int col = colBase + ni * 16 + lr;
      bf[ni].u[0] = *(const uint4*)&lb[col * LDST + kb];
      bf[ni].u[1] = *(const uint4*)&lb[col * LDST + kb + 16];
    }
#pragma unroll
    for (int mi = 0; mi < 4; ++mi)
#pragma unroll
      for (int ni = 0; ni < 4; ++ni)
        acc[mi][ni] = __builtin_amdgcn_wmma_f32_16x16x32_bf16(
            false, af[mi].v, false, bf[ni].v,
            (short)0, acc[mi][ni], false, false);
  };

  const int nk = K / BK;   // even (K = 2048 -> 64)

  issue(0, 0);             // prologue: tile 0 -> buffer 0

  for (int kt = 0; kt < nk; kt += 2) {
    // phase 0: consume buf0, stream kt+1 -> buf1 (kt+1 < nk always: nk even)
    wait_async0();         // my tile-kt pieces are in LDS
    __syncthreads();       // everyone's pieces landed; buf1 fully consumed
    issue((kt + 1) * BK, 1);
    compute(0);

    // phase 1: consume buf1, stream kt+2 -> buf0 (conditional on last step)
    wait_async0();
    __syncthreads();
    if (kt + 2 < nk) issue((kt + 2) * BK, 0);
    compute(1);
  }

  // epilogue: C layout = lane 0-15 -> N=lane, M=r ; lane 16-31 -> M=r+8
#pragma unroll
  for (int mi = 0; mi < 4; ++mi) {
#pragma unroll
    for (int ni = 0; ni < 4; ++ni) {
      const int nc = blockN + colBase + ni * 16 + lr;
      const float bv = bias[nc];
#pragma unroll
      for (int r = 0; r < 8; ++r) {
        const int m = blockM + rowBase + mi * 16 + (half ? r + 8 : r);
        const float val = acc[mi][ni][r] + bv;
        if (outF32) ((float*)out)[(size_t)m * N + nc] = val;
        else        ((__bf16*)out)[(size_t)m * N + nc] = (__bf16)val;
      }
    }
  }
}

// ---------------------------------------------------------------------------
// attention over seq = B = 4, one wave per (window n, head h)
// qkv: (B*N, 3E) bf16 ; o: (B*N, E) bf16
// ---------------------------------------------------------------------------
__global__ __launch_bounds__(256)
void attn_kernel(const __bf16* __restrict__ qkv, __bf16* __restrict__ o)
{
  const int wv   = threadIdx.x >> 5;
  const int lane = threadIdx.x & 31;
  const int p    = blockIdx.x * 8 + wv;   // 0 .. N*H-1
  const int h    = p & (NHEAD - 1);
  const int n    = p >> 4;

  const int dbase = h * HDIM + lane * 4;  // each lane owns 4 of the 128 dims

  float q[BSEQ][4], k[BSEQ][4], v[BSEQ][4];
#pragma unroll
  for (int s = 0; s < BSEQ; ++s) {
    const __bf16* row = qkv + (size_t)(s * NWIN + n) * K3E;
#pragma unroll
    for (int j = 0; j < 4; ++j) {
      q[s][j] = (float)row[dbase + j];
      k[s][j] = (float)row[EDIM + dbase + j];
      v[s][j] = (float)row[2 * EDIM + dbase + j];
    }
  }

  const float scale = 0.0883883476483184f; // 1/sqrt(128)
  float sc[BSEQ][BSEQ];
#pragma unroll
  for (int s = 0; s < BSEQ; ++s) {
#pragma unroll
    for (int tt = 0; tt < BSEQ; ++tt) {
      float d = 0.f;
#pragma unroll
      for (int j = 0; j < 4; ++j) d += q[s][j] * k[tt][j];
      // wave32 all-reduce
      for (int off = 16; off > 0; off >>= 1) d += __shfl_xor(d, off, 32);
      sc[s][tt] = d * scale;
    }
  }

  // softmax over tt, replicated across lanes
#pragma unroll
  for (int s = 0; s < BSEQ; ++s) {
    float mx = sc[s][0];
#pragma unroll
    for (int tt = 1; tt < BSEQ; ++tt) mx = fmaxf(mx, sc[s][tt]);
    float sum = 0.f;
#pragma unroll
    for (int tt = 0; tt < BSEQ; ++tt) { sc[s][tt] = __expf(sc[s][tt] - mx); sum += sc[s][tt]; }
    const float inv = 1.f / sum;
#pragma unroll
    for (int tt = 0; tt < BSEQ; ++tt) sc[s][tt] *= inv;
  }

#pragma unroll
  for (int s = 0; s < BSEQ; ++s) {
    __bf16* orow = o + (size_t)(s * NWIN + n) * EDIM + dbase;
#pragma unroll
    for (int j = 0; j < 4; ++j) {
      float os = 0.f;
#pragma unroll
      for (int tt = 0; tt < BSEQ; ++tt) os += sc[s][tt] * v[tt][j];
      orow[j] = (__bf16)os;
    }
  }
}

// ---------------------------------------------------------------------------
// launch
// ---------------------------------------------------------------------------
extern "C" void kernel_launch(void* const* d_in, const int* in_sizes, int n_in,
                              void* d_out, int out_size, void* d_ws, size_t ws_size,
                              hipStream_t stream) {
  const float* x     = (const float*)d_in[0];
  const float* in_w  = (const float*)d_in[1];
  const float* in_b  = (const float*)d_in[2];
  const float* out_w = (const float*)d_in[3];
  const float* out_b = (const float*)d_in[4];
  const float* mlp_w = (const float*)d_in[5];
  const float* mlp_b = (const float*)d_in[6];

  char* ws = (char*)d_ws;
  const size_t TOK_BYTES  = (size_t)MROWS * EDIM * 2;          //  67,108,864
  const size_t WIN_BYTES  = (size_t)K3E * EDIM * 2;            //  25,165,824
  const size_t WSQ_BYTES  = (size_t)EDIM * EDIM * 2;           //   8,388,608
  const size_t QKV_BYTES  = (size_t)MROWS * K3E * 2;           // 201,326,592
  const size_t O_BYTES    = TOK_BYTES;

  __bf16* tok   = (__bf16*)(ws);
  __bf16* wInB  = (__bf16*)(ws + TOK_BYTES);
  __bf16* wOutB = (__bf16*)(ws + TOK_BYTES + WIN_BYTES);
  __bf16* wMlpB = (__bf16*)(ws + TOK_BYTES + WIN_BYTES + WSQ_BYTES);
  __bf16* qkv   = (__bf16*)(ws + TOK_BYTES + WIN_BYTES + 2 * WSQ_BYTES);
  __bf16* oB    = (__bf16*)((char*)qkv + QKV_BYTES);
  __bf16* y1    = (__bf16*)((char*)oB + O_BYTES);
  float*  y2    = (float*)qkv;   // alias: qkv dead after attention (134MB < 201MB)

  const int threads = 256;

  // 1. weight conversion
  cvt_bf16_kernel<<<(K3E * EDIM) / threads, threads, 0, stream>>>(in_w, wInB, K3E * EDIM);
  cvt_bf16_kernel<<<(EDIM * EDIM) / threads, threads, 0, stream>>>(out_w, wOutB, EDIM * EDIM);
  cvt_bf16_kernel<<<(EDIM * EDIM) / threads, threads, 0, stream>>>(mlp_w, wMlpB, EDIM * EDIM);

  // 2. patchify (33,554,432 elements)
  patchify_kernel<<<(MROWS * EDIM) / threads, threads, 0, stream>>>(x, tok);

  // 3. qkv = tok @ Win^T + bin   (16384 x 6144 x 2048)
  gemm_bf16_wmma_kernel<<<dim3(K3E / BN, MROWS / BM), threads, 0, stream>>>(
      tok, wInB, in_b, (void*)qkv, MROWS, K3E, EDIM, 0);

  // 4. attention (65536 head-window problems, 8 waves per block)
  attn_kernel<<<(NWIN * NHEAD) / 8, threads, 0, stream>>>(qkv, oB);

  // 5. y1 = o @ Wout^T + bout   (16384 x 2048 x 2048)
  gemm_bf16_wmma_kernel<<<dim3(EDIM / BN, MROWS / BM), threads, 0, stream>>>(
      oB, wOutB, out_b, (void*)y1, MROWS, EDIM, EDIM, 0);

  // 6. y2 = y1 @ Wmlp^T + bmlp -> fp32 linear
  gemm_bf16_wmma_kernel<<<dim3(EDIM / BN, MROWS / BM), threads, 0, stream>>>(
      y1, wMlpB, mlp_b, (void*)y2, MROWS, EDIM, EDIM, 1);

  // 7. window-reverse scatter into fp32 output
  scatter_kernel<<<(MROWS * EDIM) / threads, threads, 0, stream>>>(y2, (float*)d_out);
}
